// TransLayer_3822520894036
// MI455X (gfx1250) — compile-verified, matrix operand order checked by reading, not run on hardware
//
#include <hip/hip_runtime.h>
#include <hip/hip_bf16.h>
#include <math.h>

// ---------------------------------------------------------------------------
// CDNA5 (gfx1250) Nystromformer block. All GEMMs via v_wmma_f32_16x16x32_bf16
// (bf16 A/B, f32 accumulate), wave32, one 16x64 output strip per wave.
// Fragment layouts follow CDNA5 ISA 7.12.2; the pair-wise K split
// makes each half-fragment b128-contiguous (codegen-confirmed: global_load_b128
// pairs feeding v_wmma directly).
// ---------------------------------------------------------------------------

typedef __attribute__((ext_vector_type(16))) __bf16 bf16x16;
typedef __attribute__((ext_vector_type(8)))  float  floatx8;

union FragAB { bf16x16 v; unsigned int u[8]; };

__device__ __forceinline__ unsigned pack2bf(float a, float b) {
  __bf16 lo = (__bf16)a, hi = (__bf16)b;
  unsigned short ul = __builtin_bit_cast(unsigned short, lo);
  unsigned short uh = __builtin_bit_cast(unsigned short, hi);
  return (unsigned)ul | ((unsigned)uh << 16);
}

// K offset of pair p (elements 2p,2p+1) for half-wave h.
__device__ __forceinline__ int kpair(int p, int h) {
  return (p < 4) ? (2 * p + 8 * h) : (8 + 2 * p + 8 * h);
}

// Row-major bf16 source (A: row=M; B: src stored [N,K] i.e. B = src^T).
__device__ __forceinline__ bf16x16 load_frag_bf16_rm(const __bf16* src, int row0,
                                                     int k0, int ld) {
  int lane = threadIdx.x & 31;
  int r = lane & 15, h = lane >> 4;
  const __bf16* base = src + (size_t)(row0 + r) * ld + k0;
  FragAB f;
#pragma unroll
  for (int p = 0; p < 8; ++p) f.u[p] = *(const unsigned*)(base + kpair(p, h));
  return f.v;
}

__device__ __forceinline__ floatx8 wmma_bf16(bf16x16 a, bf16x16 b, floatx8 c) {
  return __builtin_amdgcn_wmma_f32_16x16x32_bf16(false, a, false, b, (short)0, c,
                                                 false, false);
}

// 16x64 strip GEMM core: one A fragment shared by 4 B fragments per K step.
template <int KSTEPS>
__device__ __forceinline__ void gemm_strip(const __bf16* A, int rowA,
                                           const __bf16* B, int rowB, int ld,
                                           floatx8 acc[4]) {
#pragma unroll
  for (int ks = 0; ks < KSTEPS; ++ks) {
    int k0 = ks * 32;
    bf16x16 af = load_frag_bf16_rm(A, rowA, k0, ld);
#pragma unroll
    for (int t = 0; t < 4; ++t) {
      bf16x16 bfr = load_frag_bf16_rm(B, rowB + 16 * t, k0, ld);
      acc[t] = wmma_bf16(af, bfr, acc[t]);
    }
  }
}

// ---------------------------------------------------------------------------
// K0: weight fp32 -> bf16 + zero pinv scale scalars
__global__ void k_convert_w(const float* wqkv, const float* wout,
                            __bf16* wqkv_b, __bf16* wout_b, float* scales) {
  int i = blockIdx.x * blockDim.x + threadIdx.x;
  if (i < 2) scales[i] = 0.f;
  if (i < 1536 * 512) wqkv_b[i] = (__bf16)wqkv[i];
  if (i < 512 * 512) wout_b[i] = (__bf16)wout[i];
}

// K1: LayerNorm, one wave per token
__global__ void k_layernorm(const float* x, const float* g, const float* b,
                            __bf16* xn, int ntok) {
  int wave = (int)((blockIdx.x * blockDim.x + threadIdx.x) >> 5);
  int lane = threadIdx.x & 31;
  if (wave >= ntok) return;
  const float* xr = x + (size_t)wave * 512;
  float s = 0.f, s2 = 0.f, vals[16];
#pragma unroll
  for (int i = 0; i < 16; ++i) {
    float v = xr[lane + 32 * i];
    vals[i] = v; s += v; s2 += v * v;
  }
  for (int m = 16; m >= 1; m >>= 1) {
    s += __shfl_xor(s, m, 32);
    s2 += __shfl_xor(s2, m, 32);
  }
  float mu = s * (1.f / 512.f);
  float var = s2 * (1.f / 512.f) - mu * mu;
  float inv = rsqrtf(var + 1e-5f);
  __bf16* o = xn + (size_t)wave * 512;
#pragma unroll
  for (int i = 0; i < 16; ++i) {
    int c = lane + 32 * i;
    o[c] = (__bf16)((vals[i] - mu) * inv * g[c] + b[c]);
  }
}

// K2: QKV GEMM [16384,512]x[512,1536]; 16x64 strip per wave (one head/strip)
__global__ void k_gemm_qkv(const __bf16* xn, const __bf16* wq, __bf16* qb,
                           __bf16* kb, float* vf, __bf16* vT) {
  int wid = (int)((blockIdx.x * blockDim.x + threadIdx.x) >> 5);  // 1024*24
  int tM = wid / 24, s = wid % 24;
  floatx8 acc[4] = {};
  gemm_strip<16>(xn, tM * 16, wq, s * 64, 512, acc);
  int lane = threadIdx.x & 31;
  int n = lane & 15, h = lane >> 4;
  int col0 = s * 64;
  int sect = col0 >> 9;              // 0=q 1=k 2=v
  int head = (col0 >> 6) & 7;        // strip == exactly one head
#pragma unroll
  for (int t = 0; t < 4; ++t) {
    int dh = 16 * t + n;
#pragma unroll
    for (int r = 0; r < 8; ++r) {
      int tok = tM * 16 + r + 8 * h;
      int bb = tok >> 12, ns = tok & 4095;
      size_t bh = (size_t)(bb * 8 + head);
      size_t idx = (bh * 4096 + ns) * 64 + dh;
      float val = acc[t][r];
      if (sect == 0) qb[idx] = (__bf16)(val * 0.125f);   // q * dh^-0.5
      else if (sect == 1) kb[idx] = (__bf16)val;
      else {
        vf[idx] = val;
        vT[(bh * 64 + dh) * 4096 + ns] = (__bf16)val;
      }
    }
  }
}

// K3: landmark means (l=16)
__global__ void k_landmarks(const __bf16* qb, const __bf16* kb, __bf16* ql,
                            __bf16* kl) {
  int i = blockIdx.x * blockDim.x + threadIdx.x;  // 32*256*64
  if (i >= 32 * 256 * 64) return;
  int dh = i & 63, m = (i >> 6) & 255, bh = i >> 14;
  size_t base = ((size_t)bh * 4096 + m * 16) * 64 + dh;
  float sq = 0.f, sk = 0.f;
#pragma unroll
  for (int t = 0; t < 16; ++t) {
    sq += (float)qb[base + t * 64];
    sk += (float)kb[base + t * 64];
  }
  ql[i] = (__bf16)(sq * (1.f / 16.f));
  kl[i] = (__bf16)(sk * (1.f / 16.f));
}

// K4: attn2 logits = q_l @ k_l^T [32,256,256] f32
__global__ void k_attn2_logits(const __bf16* ql, const __bf16* kl, float* out) {
  int wid = (int)((blockIdx.x * blockDim.x + threadIdx.x) >> 5);  // 32*16*4
  int bh = wid >> 6, t = wid & 63;
  int tM = t >> 2, s = t & 3;
  const __bf16* A = ql + (size_t)bh * 256 * 64;
  const __bf16* B = kl + (size_t)bh * 256 * 64;
  floatx8 acc[4] = {};
  gemm_strip<2>(A, tM * 16, B, s * 64, 64, acc);
  int lane = threadIdx.x & 31;
  int n = lane & 15, h = lane >> 4;
  float* o = out + (size_t)bh * 256 * 256;
#pragma unroll
  for (int t4 = 0; t4 < 4; ++t4)
#pragma unroll
    for (int r = 0; r < 8; ++r)
      o[(size_t)(tM * 16 + r + 8 * h) * 256 + s * 64 + 16 * t4 + n] =
          acc[t4][r];
}

// K5: row softmax (256) in place + bf16 copy (pinv A operand)
__global__ void k_softmax256(float* x, __bf16* xb, int nrows) {
  int wave = (int)((blockIdx.x * blockDim.x + threadIdx.x) >> 5);
  if (wave >= nrows) return;
  int lane = threadIdx.x & 31;
  float* r = x + (size_t)wave * 256;
  __bf16* rb = xb + (size_t)wave * 256;
  float v[8], m = -1e30f;
#pragma unroll
  for (int i = 0; i < 8; ++i) { v[i] = r[lane + 32 * i]; m = fmaxf(m, v[i]); }
  for (int t = 16; t >= 1; t >>= 1) m = fmaxf(m, __shfl_xor(m, t, 32));
  float s = 0.f;
#pragma unroll
  for (int i = 0; i < 8; ++i) { v[i] = __expf(v[i] - m); s += v[i]; }
  for (int t = 16; t >= 1; t >>= 1) s += __shfl_xor(s, t, 32);
  float inv = 1.f / s;
#pragma unroll
  for (int i = 0; i < 8; ++i) {
    float pv = v[i] * inv;
    r[lane + 32 * i] = pv;
    rb[lane + 32 * i] = (__bf16)pv;
  }
}

// K6: global max of row-sums (scales[0]) / col-sums (scales[1]) of |attn2|
__global__ void k_pinv_scales(const float* x, float* scales) {
  int wave = (int)((blockIdx.x * blockDim.x + threadIdx.x) >> 5);  // 32*256
  if (wave >= 32 * 256) return;
  int lane = threadIdx.x & 31;
  int bh = wave >> 8, r = wave & 255;
  const float* m = x + (size_t)bh * 256 * 256;
  float rs = 0.f, cs = 0.f;
#pragma unroll
  for (int i = 0; i < 8; ++i) {
    rs += fabsf(m[(size_t)r * 256 + lane + 32 * i]);
    cs += fabsf(m[(size_t)(lane + 32 * i) * 256 + r]);
  }
  for (int t = 16; t >= 1; t >>= 1) {
    rs += __shfl_xor(rs, t, 32);
    cs += __shfl_xor(cs, t, 32);
  }
  if (lane == 0) {  // positive floats: uint compare == float compare
    atomicMax((unsigned*)&scales[0], __float_as_uint(rs));
    atomicMax((unsigned*)&scales[1], __float_as_uint(cs));
  }
}

// K7: z = x^T/(col*row) -> bf16 z (row-major) + z^T (B-operand layout)
__global__ void k_init_z(const float* x, __bf16* zb, __bf16* zTb,
                         const float* scales) {
  size_t i = (size_t)blockIdx.x * blockDim.x + threadIdx.x;
  if (i >= (size_t)32 * 256 * 256) return;
  int j = (int)(i & 255), r = (int)((i >> 8) & 255);  // i = [bh][M=r][N=j]
  size_t bh = i >> 16;
  float denom = scales[0] * scales[1];
  float val = x[(bh << 16) + (size_t)j * 256 + r] / denom;  // x^T
  zb[i] = (__bf16)val;
  zTb[(bh << 16) + (size_t)j * 256 + r] = (__bf16)val;      // [N][M]
}

// K8: bf16-resident batched 256^3 GEMM, 16x64 strip per wave.
// T = A @ BT^T. Optional: OutRaw = T (rm), OutT/OutTT = alpha*I+beta*T (rm/tr).
__global__ void k_gemm256b(const __bf16* A, const __bf16* BT, __bf16* OutRaw,
                           __bf16* OutT, __bf16* OutTT, float alpha,
                           float beta) {
  int wid = (int)((blockIdx.x * blockDim.x + threadIdx.x) >> 5);  // 32*16*4
  int bh = wid >> 6, t = wid & 63;
  int tM = t >> 2, s = t & 3;
  size_t bofs = (size_t)bh << 16;
  floatx8 acc[4] = {};
  gemm_strip<8>(A + bofs, tM * 16, BT + bofs, s * 64, 256, acc);
  int lane = threadIdx.x & 31;
  int n = lane & 15, h = lane >> 4;
#pragma unroll
  for (int t4 = 0; t4 < 4; ++t4)
#pragma unroll
    for (int r = 0; r < 8; ++r) {
      int M = tM * 16 + r + 8 * h, N = s * 64 + 16 * t4 + n;
      float tv = acc[t4][r];
      if (OutRaw) OutRaw[bofs + (size_t)M * 256 + N] = (__bf16)tv;
      float tr = alpha * (M == N ? 1.f : 0.f) + beta * tv;
      if (OutT) OutT[bofs + (size_t)M * 256 + N] = (__bf16)tr;
      if (OutTT) OutTT[bofs + (size_t)N * 256 + M] = (__bf16)tr;
    }
}

// ---------------------------------------------------------------------------
// Fused streaming softmax-attention (transposed-product trick): the f32
// D-fragment of S^T pair-packs directly into the bf16 B-fragment for
// O^T += A2[64,K] @ P^T -- no cross-lane movement.
__device__ __forceinline__ void stream_attn(const __bf16* Ak, const __bf16* Bq,
                                            int qrow0, const __bf16* A2,
                                            int ldA2, int nkeys, float* outF,
                                            int rsF, __bf16* outB, int nsB,
                                            int dsB) {
  int lane = threadIdx.x & 31;
  int n = lane & 15, h = lane >> 4;
  bf16x16 bq0 = load_frag_bf16_rm(Bq, qrow0, 0, 64);
  bf16x16 bq1 = load_frag_bf16_rm(Bq, qrow0, 32, 64);
  float run_m = -1e30f, run_s = 0.f;
  floatx8 acc[4] = {};
  for (int c = 0; c < nkeys; c += 32) {
    floatx8 s0 = {}, s1 = {};
    s0 = wmma_bf16(load_frag_bf16_rm(Ak, c, 0, 64), bq0, s0);
    s0 = wmma_bf16(load_frag_bf16_rm(Ak, c, 32, 64), bq1, s0);
    s1 = wmma_bf16(load_frag_bf16_rm(Ak, c + 16, 0, 64), bq0, s1);
    s1 = wmma_bf16(load_frag_bf16_rm(Ak, c + 16, 32, 64), bq1, s1);
    float mc = -1e30f;
#pragma unroll
    for (int r = 0; r < 8; ++r) { mc = fmaxf(mc, s0[r]); mc = fmaxf(mc, s1[r]); }
    mc = fmaxf(mc, __shfl_xor(mc, 16, 32));  // merge column halves
    float newm = fmaxf(run_m, mc);
    float scale = __expf(run_m - newm);
    float p0[8], p1[8], ps = 0.f;
#pragma unroll
    for (int r = 0; r < 8; ++r) {
      p0[r] = __expf(s0[r] - newm);
      p1[r] = __expf(s1[r] - newm);
      ps += p0[r] + p1[r];
    }
    run_s = run_s * scale + ps;
    run_m = newm;
    FragAB bp;  // D-frag pairs == bf16 B-frag K split per half
#pragma unroll
    for (int p = 0; p < 4; ++p) {
      bp.u[p] = pack2bf(p0[2 * p], p0[2 * p + 1]);
      bp.u[4 + p] = pack2bf(p1[2 * p], p1[2 * p + 1]);
    }
#pragma unroll
    for (int t = 0; t < 4; ++t) {
#pragma unroll
      for (int r = 0; r < 8; ++r) acc[t][r] *= scale;
      acc[t] = wmma_bf16(load_frag_bf16_rm(A2, 16 * t, c, ldA2), bp.v, acc[t]);
    }
  }
  float tot = run_s + __shfl_xor(run_s, 16, 32);
  float inv = 1.f / tot;
#pragma unroll
  for (int t = 0; t < 4; ++t)
#pragma unroll
    for (int r = 0; r < 8; ++r) {
      int dh = 16 * t + r + 8 * h;
      float val = acc[t][r] * inv;
      if (outF) outF[(size_t)n * rsF + dh] = val;
      if (outB) outB[(size_t)n * nsB + (size_t)dh * dsB] = (__bf16)val;
    }
}

// K9: kv^T = (softmax(q_l @ k^T) @ v)^T as bf16 [bh,64,256]
__global__ void k_attn3v(const __bf16* kb, const __bf16* ql, const __bf16* vT,
                         __bf16* kvT) {
  int wid = (int)((blockIdx.x * blockDim.x + threadIdx.x) >> 5);  // 32*16
  int bh = wid >> 4, blk = wid & 15;
  stream_attn(kb + (size_t)bh * 4096 * 64, ql + (size_t)bh * 256 * 64, blk * 16,
              vT + (size_t)bh * 64 * 4096, 4096, 4096, nullptr, 0,
              kvT + (size_t)bh * 64 * 256 + blk * 16, 1, 256);
}

// K10: W2 = attn2_inv @ kv (bf16 b128 GEMM), stored transposed [bh,64,256]
__global__ void k_w2(const __bf16* zb, const __bf16* kvT, __bf16* w2t) {
  int wid = (int)((blockIdx.x * blockDim.x + threadIdx.x) >> 5);  // 32*16
  int bh = wid >> 4, tM = wid & 15;
  floatx8 acc[4] = {};
  gemm_strip<8>(zb + ((size_t)bh << 16), tM * 16, kvT + (size_t)bh * 64 * 256,
                0, 256, acc);
  int lane = threadIdx.x & 31;
  int n = lane & 15, h = lane >> 4;
#pragma unroll
  for (int t4 = 0; t4 < 4; ++t4)
#pragma unroll
    for (int r = 0; r < 8; ++r) {
      int M = tM * 16 + r + 8 * h;  // landmark
      int N = 16 * t4 + n;          // dh
      w2t[((size_t)bh * 64 + N) * 256 + M] = (__bf16)acc[t4][r];
    }
}

// K11: out = softmax(q @ k_l^T) @ W2, streamed over 256 landmarks
__global__ void k_attn1_out(const __bf16* kl, const __bf16* qb,
                            const __bf16* w2t, float* aout) {
  int wid = (int)((blockIdx.x * blockDim.x + threadIdx.x) >> 5);  // 32*256
  int bh = wid >> 8, blk = wid & 255;
  int b = bh >> 3, hh = bh & 7;
  stream_attn(kl + (size_t)bh * 256 * 64, qb + (size_t)bh * 4096 * 64, blk * 16,
              w2t + (size_t)bh * 64 * 256, 256, 256,
              aout + ((size_t)b * 4096 + blk * 16) * 512 + hh * 64, 512,
              nullptr, 0, 0);
}

// K12: depthwise conv residual over sequence + bf16 copy of attn_out
__global__ void k_conv_res(const float* vf, const float* rw, float* aout,
                           __bf16* aout_b) {
  size_t i = (size_t)blockIdx.x * blockDim.x + threadIdx.x;
  if (i >= (size_t)4 * 4096 * 512) return;
  int col = (int)(i & 511);
  size_t tok = i >> 9;
  int b = (int)(tok >> 12), ns = (int)(tok & 4095);
  int hh = col >> 6, dh = col & 63;
  const float* vrow = vf + ((size_t)(b * 8 + hh) * 4096) * 64 + dh;
  float s = 0.f;
#pragma unroll
  for (int j = 0; j < 33; ++j) {
    int p = ns + j - 16;
    if (p >= 0 && p < 4096) s += vrow[(size_t)p * 64] * rw[hh * 33 + j];
  }
  float val = aout[i] + s;
  aout[i] = val;
  aout_b[i] = (__bf16)val;
}

// K13: final = x + attn_out @ w_out^T + b_out; 16x64 strip per wave
__global__ void k_outproj(const __bf16* aoutb, const __bf16* wo,
                          const float* bias, const float* x, float* out) {
  int wid = (int)((blockIdx.x * blockDim.x + threadIdx.x) >> 5);  // 1024*8
  int tM = wid >> 3, s = wid & 7;
  floatx8 acc[4] = {};
  gemm_strip<16>(aoutb, tM * 16, wo, s * 64, 512, acc);
  int lane = threadIdx.x & 31;
  int n = lane & 15, h = lane >> 4;
#pragma unroll
  for (int t4 = 0; t4 < 4; ++t4) {
    int col = s * 64 + 16 * t4 + n;
    float bc = bias[col];
#pragma unroll
    for (int r = 0; r < 8; ++r) {
      size_t idx = (size_t)(tM * 16 + r + 8 * h) * 512 + col;
      out[idx] = x[idx] + acc[t4][r] + bc;
    }
  }
}

// ---------------------------------------------------------------------------
extern "C" void kernel_launch(void* const* d_in, const int* in_sizes, int n_in,
                              void* d_out, int out_size, void* d_ws,
                              size_t ws_size, hipStream_t stream) {
  const float* x     = (const float*)d_in[0];
  const float* ln_g  = (const float*)d_in[1];
  const float* ln_b  = (const float*)d_in[2];
  const float* w_qkv = (const float*)d_in[3];
  const float* w_out = (const float*)d_in[4];
  const float* b_out = (const float*)d_in[5];
  const float* res_w = (const float*)d_in[6];
  float* out = (float*)d_out;

  char* p = (char*)d_ws;
  auto alloc = [&](size_t bytes) -> void* {
    void* r = (void*)p;
    p += (bytes + 255) & ~(size_t)255;
    return r;
  };
  const size_t NTOK = 4 * 4096;
  const size_t MM = (size_t)32 * 256 * 256;  // pinv matrix elems
  __bf16* xn_b   = (__bf16*)alloc(NTOK * 512 * 2);
  __bf16* wqkv_b = (__bf16*)alloc(1536 * 512 * 2);
  __bf16* wout_b = (__bf16*)alloc(512 * 512 * 2);
  __bf16* q_b    = (__bf16*)alloc((size_t)32 * 4096 * 64 * 2);
  __bf16* k_b    = (__bf16*)alloc((size_t)32 * 4096 * 64 * 2);
  float*  v_f    = (float*)alloc((size_t)32 * 4096 * 64 * 4);
  __bf16* vT_b   = (__bf16*)alloc((size_t)32 * 64 * 4096 * 2);
  __bf16* ql_b   = (__bf16*)alloc((size_t)32 * 256 * 64 * 2);
  __bf16* kl_b   = (__bf16*)alloc((size_t)32 * 256 * 64 * 2);
  float*  mat_x  = (float*)alloc(MM * 4);    // f32 logits/softmax (scales)
  __bf16* x_b    = (__bf16*)alloc(MM * 2);   // bf16 softmax (pinv A)
  __bf16* z0_b   = (__bf16*)alloc(MM * 2);
  __bf16* z0T_b  = (__bf16*)alloc(MM * 2);
  __bf16* z1_b   = (__bf16*)alloc(MM * 2);
  __bf16* z1T_b  = (__bf16*)alloc(MM * 2);
  __bf16* xz_b   = (__bf16*)alloc(MM * 2);
  __bf16* t1T_b  = (__bf16*)alloc(MM * 2);
  __bf16* t2T_b  = (__bf16*)alloc(MM * 2);
  __bf16* t3T_b  = (__bf16*)alloc(MM * 2);
  __bf16* kvT_b  = (__bf16*)alloc((size_t)32 * 64 * 256 * 2);
  __bf16* w2t_b  = (__bf16*)alloc((size_t)32 * 64 * 256 * 2);
  float*  aout_f = (float*)alloc(NTOK * 512 * 4);
  __bf16* aout_b = (__bf16*)alloc(NTOK * 512 * 2);
  float*  scales = (float*)alloc(256);
  (void)ws_size; (void)n_in; (void)in_sizes; (void)out_size;

  const int T = 256;  // 8 waves / workgroup
  k_convert_w<<<3072, T, 0, stream>>>(w_qkv, w_out, wqkv_b, wout_b, scales);
  k_layernorm<<<2048, T, 0, stream>>>(x, ln_g, ln_b, xn_b, (int)NTOK);
  k_gemm_qkv<<<3072, T, 0, stream>>>(xn_b, wqkv_b, q_b, k_b, v_f, vT_b);
  k_landmarks<<<2048, T, 0, stream>>>(q_b, k_b, ql_b, kl_b);
  k_attn2_logits<<<256, T, 0, stream>>>(ql_b, kl_b, mat_x);
  k_softmax256<<<1024, T, 0, stream>>>(mat_x, x_b, 32 * 256);
  k_pinv_scales<<<1024, T, 0, stream>>>(mat_x, scales);
  k_init_z<<<8192, T, 0, stream>>>(mat_x, z0_b, z0T_b, scales);

  // Newton-Schulz pinv: z = 0.25*z@(13I - xz@(15I - xz@(7I - xz)))
  // bf16-resident: every operand pre-laid-out for row-major b128 frag loads.
  __bf16* zc = z0_b;  __bf16* zcT = z0T_b;
  __bf16* zn = z1_b;  __bf16* znT = z1T_b;
  for (int it = 0; it < 6; ++it) {
    k_gemm256b<<<256, T, 0, stream>>>(x_b, zcT, xz_b, nullptr, t1T_b, 7.f, -1.f);
    k_gemm256b<<<256, T, 0, stream>>>(xz_b, t1T_b, nullptr, nullptr, t2T_b, 15.f, -1.f);
    k_gemm256b<<<256, T, 0, stream>>>(xz_b, t2T_b, nullptr, nullptr, t3T_b, 13.f, -1.f);
    k_gemm256b<<<256, T, 0, stream>>>(zc, t3T_b, nullptr, zn, znT, 0.f, 0.25f);
    __bf16* t;
    t = zc; zc = zn; zn = t;
    t = zcT; zcT = znT; znT = t;
  }

  k_attn3v<<<64, T, 0, stream>>>(k_b, ql_b, vT_b, kvT_b);
  k_w2<<<64, T, 0, stream>>>(zc, kvT_b, w2t_b);
  k_attn1_out<<<1024, T, 0, stream>>>(kl_b, q_b, w2t_b, aout_f);
  k_conv_res<<<32768, T, 0, stream>>>(v_f, res_w, aout_f, aout_b);
  k_outproj<<<1024, T, 0, stream>>>(aout_b, wout_b, b_out, x, out);
}
